// GNNDecoder_73598559584324
// MI455X (gfx1250) — matrix operand order repaired and checked by reading, not compiled, not run on hardware
//
#include <hip/hip_runtime.h>

typedef _Float16 v2h  __attribute__((ext_vector_type(2)));
typedef _Float16 v8h  __attribute__((ext_vector_type(8)));
typedef _Float16 v16h __attribute__((ext_vector_type(16)));
typedef float    v8f  __attribute__((ext_vector_type(8)));

#define DIM     128
#define D2      256
#define NPAIRS  500000
#define MBLK    128
#define NBLKS   ((NPAIRS + MBLK - 1) / MBLK)   // 3907
#define ROWH    264                            // halfs per padded row (512B + 16B pad)
#define ROWU    132                            // dwords per padded row
#define W1T_BYTES (D2 * ROWH * 2)              // 135168
#define A_BYTES   (MBLK * ROWH * 2)            // 67584
#define PRED_OFF  (W1T_BYTES + A_BYTES)        // 202752
#define SMEM_BYTES (PRED_OFF + MBLK * 4)       // 203264

static __device__ __forceinline__ unsigned pack_f16x2(float f0, float f1) {
    v2h pk;
    pk[0] = (_Float16)f0;
    pk[1] = (_Float16)f1;
    return __builtin_bit_cast(unsigned, pk);
}

extern "C" __global__ __launch_bounds__(256, 1)
void gnn_decoder_wmma(const float* __restrict__ x,
                      const long long* __restrict__ junc,
                      const float* __restrict__ W1,
                      const float* __restrict__ b1,
                      const float* __restrict__ W2,
                      const float* __restrict__ b2,
                      float* __restrict__ out)
{
    extern __shared__ char smem[];
    _Float16* w1t   = (_Float16*)smem;                 // W1^T, f16, [n][k], padded rows
    unsigned* w1t32 = (unsigned*)smem;
    _Float16* aT    = (_Float16*)(smem + W1T_BYTES);   // gathered concat rows, f16
    unsigned* a32   = (unsigned*)(smem + W1T_BYTES);
    float*    pred  = (float*)(smem + PRED_OFF);       // per-pair partial sums

    const int tid  = threadIdx.x;
    const int lane = tid & 31;
    const int wv   = tid >> 5;      // 0..7
    const int jg   = wv & 3;        // column group: 4 j-tiles (64 cols)
    const int mg   = wv >> 2;       // row group: 4 m-tiles (64 rows)
    const int g    = lane >> 4;     // half-wave
    const int lm   = lane & 15;

    // ---- stage W1^T as f16 into LDS (once per persistent workgroup) ----
    for (int it = 0; it < 128; ++it) {
        int idx = it * 256 + tid;
        int n   = idx & 255;        // output feature
        int kp  = idx >> 8;         // packed pair of consecutive k (0..127)
        float f0 = W1[(2 * kp    ) * D2 + n];
        float f1 = W1[(2 * kp + 1) * D2 + n];
        w1t32[n * ROWU + kp] = pack_f16x2(f0, f1);
    }

    // per-lane layer-2 constants: column n is fixed per (wave, lane) for the whole kernel
    float b1v[4], w2v[4];
#pragma unroll
    for (int jj = 0; jj < 4; ++jj) {
        int n = (4 * jg + jj) * 16 + lm;
        b1v[jj] = b1[n];
        w2v[jj] = W2[n];
    }
    const float b2v = b2[0];

    __syncthreads();

    for (int blk = blockIdx.x; blk < NBLKS; blk += gridDim.x) {
        const long long pbase = (long long)blk * MBLK;

        if (tid < MBLK) pred[tid] = 0.0f;

        // ---- gather + f16-convert the 128 concat rows into LDS ----
        for (int it = 0; it < 64; ++it) {
            int idx = it * 256 + tid;
            int kp  = idx & 127;            // packed k-pair within the 256-row
            int r   = idx >> 7;             // 0..127 local pair
            long long p = pbase + r;
            long long j0 = 0, j1 = 0;
            if (p < NPAIRS) { j0 = junc[2 * p]; j1 = junc[2 * p + 1]; }
            long long node = (kp < 64) ? j0 : j1;
            int ko = (kp & 63) * 2;
            const float* src = x + node * DIM + ko;
            a32[r * ROWU + kp] = pack_f16x2(src[0], src[1]);
        }
        __syncthreads();

        // ---- layer 1: 128x256x256 GEMM tile via v_wmma_f32_16x16x32_f16 ----
        v8f acc[4][4];
#pragma unroll
        for (int mi = 0; mi < 4; ++mi)
#pragma unroll
            for (int jj = 0; jj < 4; ++jj)
#pragma unroll
                for (int v = 0; v < 8; ++v)
                    acc[mi][jj][v] = 0.0f;

        for (int s = 0; s < 8; ++s) {       // K steps of 32
            v16h bfr[4];
#pragma unroll
            for (int jj = 0; jj < 4; ++jj) {
                int n = (4 * jg + jj) * 16 + lm;
                const v8h* bp = (const v8h*)(w1t + n * ROWH + s * 32 + 8 * g);
                v8h lo = bp[0];
                v8h hi = bp[2];
                bfr[jj] = __builtin_shufflevector(lo, hi,
                    0,1,2,3,4,5,6,7,8,9,10,11,12,13,14,15);
            }
#pragma unroll
            for (int mi = 0; mi < 4; ++mi) {
                int row = (mg * 4 + mi) * 16 + lm;
                const _Float16* ap = aT + row * ROWH + s * 32 + 8 * g;
                v8h lo = *(const v8h*)ap;
                v8h hi = *(const v8h*)(ap + 16);
                v16h afr = __builtin_shufflevector(lo, hi,
                    0,1,2,3,4,5,6,7,8,9,10,11,12,13,14,15);
#pragma unroll
                for (int jj = 0; jj < 4; ++jj) {
                    acc[mi][jj] = __builtin_amdgcn_wmma_f32_16x16x32_f16(
                        false, afr, false, bfr[jj],
                        (short)0, acc[mi][jj], false, false);
                }
            }
        }

        // ---- fused layer 2: relu(acc + b1) . W2, reduce across columns ----
#pragma unroll
        for (int mi = 0; mi < 4; ++mi) {
            float ps[8];
#pragma unroll
            for (int v = 0; v < 8; ++v) ps[v] = 0.0f;
#pragma unroll
            for (int jj = 0; jj < 4; ++jj)
#pragma unroll
                for (int v = 0; v < 8; ++v) {
                    float t = acc[mi][jj][v] + b1v[jj];
                    t = fmaxf(t, 0.0f);
                    ps[v] = __builtin_fmaf(t, w2v[jj], ps[v]);
                }
            // reduce over the 16-lane column group (lane bits 0..3)
#pragma unroll
            for (int v = 0; v < 8; ++v) {
                float t = ps[v];
                t += __shfl_xor(t, 1, 32);
                t += __shfl_xor(t, 2, 32);
                t += __shfl_xor(t, 4, 32);
                t += __shfl_xor(t, 8, 32);
                ps[v] = t;
            }
            if (lm == 0) {
#pragma unroll
                for (int v = 0; v < 8; ++v) {
                    int row = (mg * 4 + mi) * 16 + v + 8 * g;
                    atomicAdd(&pred[row], ps[v]);   // ds_add_f32 across the 4 jg waves
                }
            }
        }
        __syncthreads();

        // ---- store ----
        if (tid < MBLK) {
            long long p = pbase + tid;
            if (p < NPAIRS) out[p] = pred[tid] + b2v;
        }
        __syncthreads();    // protect pred / A reuse next iteration
    }
}

extern "C" void kernel_launch(void* const* d_in, const int* in_sizes, int n_in,
                              void* d_out, int out_size, void* d_ws, size_t ws_size,
                              hipStream_t stream) {
    (void)in_sizes; (void)n_in; (void)d_ws; (void)ws_size; (void)out_size;
    const float*     x    = (const float*)d_in[0];
    const long long* junc = (const long long*)d_in[1];
    const float*     W1   = (const float*)d_in[2];
    const float*     b1   = (const float*)d_in[3];
    const float*     W2   = (const float*)d_in[4];
    const float*     b2   = (const float*)d_in[5];
    float*           out  = (float*)d_out;

    // allow > 64 KB dynamic LDS (203 KB of the WGP's 320 KB)
    (void)hipFuncSetAttribute((const void*)gnn_decoder_wmma,
                              hipFuncAttributeMaxDynamicSharedMemorySize, SMEM_BYTES);

    gnn_decoder_wmma<<<1024, 256, SMEM_BYTES, stream>>>(x, junc, W1, b1, W2, b2, out);
}
